// ParallelMultiHeadedAttention_28415503631168
// MI455X (gfx1250) — matrix operand order repaired and checked by reading, not compiled, not run on hardware
//
#include <hip/hip_runtime.h>
#include <hip/hip_bf16.h>
#include <math.h>

// ---------------------------------------------------------------------------
// MHA for MI455X (gfx1250): bf16 WMMA (v_wmma_f32_16x16x32_bf16) everywhere,
// flash-style online softmax so the 2x16x2048x2048 score tensor never touches
// HBM. position_bias (256MB) streamed once, fused into score tiles.
// Async global->LDS staging (GLOBAL_LOAD_ASYNC_TO_LDS_B128, ASYNCcnt) for the
// attention K tiles and the out-proj A tiles; this toolchain's builtin takes
// (int4* src, int4* dst, imm offset, imm cpol).
// ---------------------------------------------------------------------------

#define HIDDEN 2048
#define HEADS 16
#define HD 128
#define INNER 2048
#define BATCH 2
#define SEQ 2048

typedef __attribute__((ext_vector_type(16))) __bf16 v16bf;
typedef __attribute__((ext_vector_type(8)))  float v8f;
typedef __attribute__((ext_vector_type(4)))  int   v4i_t;

union FragAB { v16bf v; unsigned short u[16]; };
union FragC  { v8f   v; float          f[8];  };

#if defined(__has_builtin)
#if __has_builtin(__builtin_amdgcn_global_load_async_to_lds_b128)
#define HAVE_ASYNC_LDS 1
#endif
#endif

__device__ __forceinline__ void async_copy_b128(const void* g, void* l) {
#ifdef HAVE_ASYNC_LDS
    __builtin_amdgcn_global_load_async_to_lds_b128(
        (v4i_t*)(void*)g, (v4i_t*)l, 0, 0);
#else
    *(uint4*)l = *(const uint4*)g;
#endif
}

__device__ __forceinline__ void async_wait() {
#ifdef HAVE_ASYNC_LDS
#if __has_builtin(__builtin_amdgcn_s_wait_asynccnt)
    __builtin_amdgcn_s_wait_asynccnt(0);
#else
    asm volatile("s_wait_asynccnt 0" ::: "memory");
#endif
#endif
}

__device__ __forceinline__ unsigned short f2bf(float f) {
    unsigned int x = __float_as_uint(f);
    // round-to-nearest-even truncation to bf16
    unsigned int r = (x + 0x7FFFu + ((x >> 16) & 1u)) >> 16;
    return (unsigned short)r;
}

// ===========================================================================
// Kernel 1: fused QKV projection.  out = X @ W + b   (cast to bf16)
// blockIdx.z selects (query,Wq,bq)->Q / (key,Wk,bk)->K / (value,Wv,bv)->V
// Block tile 128x128, 8 waves, each wave 32x64 = 2x4 WMMA tiles, K-step 32.
// Output layout: [b, h, s, d] bf16 for the attention kernel.
// ===========================================================================
__global__ __launch_bounds__(256) void qkv_proj_kernel(
    const float* __restrict__ q, const float* __restrict__ k, const float* __restrict__ v,
    const float* __restrict__ Wq, const float* __restrict__ Wk, const float* __restrict__ Wv,
    const float* __restrict__ bq, const float* __restrict__ bk, const float* __restrict__ bv,
    unsigned short* __restrict__ Qbf, unsigned short* __restrict__ Kbf,
    unsigned short* __restrict__ Vbf)
{
    const int proj = blockIdx.z;
    const float* X  = (proj == 0) ? q  : (proj == 1) ? k  : v;
    const float* W  = (proj == 0) ? Wq : (proj == 1) ? Wk : Wv;
    const float* Bv = (proj == 0) ? bq : (proj == 1) ? bk : bv;
    unsigned short* O = (proj == 0) ? Qbf : (proj == 1) ? Kbf : Vbf;

    __shared__ __align__(16) unsigned short ldsA[128 * 40];   // [m][k], bf16
    __shared__ __align__(16) unsigned short ldsB[128 * 40];   // transposed: [n][k]

    const int tid  = threadIdx.x;
    const int wid  = tid >> 5;
    const int lane = tid & 31;
    const int half = lane >> 4;
    const int l15  = lane & 15;

    const int bm = blockIdx.y * 128;
    const int bn = blockIdx.x * 128;
    const int wm = (wid & 3) * 32;
    const int wn = (wid >> 2) * 64;

    FragC acc[2][4];
#pragma unroll
    for (int tm = 0; tm < 2; tm++)
#pragma unroll
        for (int tn = 0; tn < 4; tn++)
#pragma unroll
            for (int e = 0; e < 8; e++) acc[tm][tn].f[e] = 0.0f;

    for (int k0 = 0; k0 < HIDDEN; k0 += 32) {
        // stage A tile 128x32 (f32 -> bf16)
#pragma unroll
        for (int t = 0; t < 16; t++) {
            int idx = t * 256 + tid;
            int r = idx >> 5, c = idx & 31;
            ldsA[r * 40 + c] = f2bf(X[(size_t)(bm + r) * HIDDEN + k0 + c]);
        }
        // stage B tile 32x128 transposed (f32 -> bf16)
#pragma unroll
        for (int t = 0; t < 16; t++) {
            int idx = t * 256 + tid;
            int r = idx >> 7, c = idx & 127;
            ldsB[c * 40 + r] = f2bf(W[(size_t)(k0 + r) * INNER + bn + c]);
        }
        __syncthreads();

        FragAB a[2], b[4];
#pragma unroll
        for (int tm = 0; tm < 2; tm++) {
            int mrow = wm + tm * 16 + l15;
#pragma unroll
            for (int i = 0; i < 16; i++) {
                int kk = half * 16 + i;
                a[tm].u[i] = ldsA[mrow * 40 + kk];
            }
        }
#pragma unroll
        for (int tn = 0; tn < 4; tn++) {
            int ncol = wn + tn * 16 + l15;
#pragma unroll
            for (int i = 0; i < 16; i++)
                b[tn].u[i] = ldsB[ncol * 40 + half * 16 + i];
        }
#pragma unroll
        for (int tm = 0; tm < 2; tm++)
#pragma unroll
            for (int tn = 0; tn < 4; tn++)
                acc[tm][tn].v = __builtin_amdgcn_wmma_f32_16x16x32_bf16(
                    false, a[tm].v, false, b[tn].v, (short)0, acc[tm][tn].v,
                    false, false);
        __syncthreads();
    }

    // epilogue: bias add, bf16 cast, scatter to [b,h,s,d]
#pragma unroll
    for (int tm = 0; tm < 2; tm++)
#pragma unroll
        for (int tn = 0; tn < 4; tn++)
#pragma unroll
            for (int e = 0; e < 8; e++) {
                int row = bm + wm + tm * 16 + e + 8 * half;
                int col = bn + wn + tn * 16 + l15;
                float val = acc[tm][tn].f[e] + Bv[col];
                int bb = row >> 11, s = row & (SEQ - 1);
                int h  = col >> 7,  d = col & (HD - 1);
                O[((size_t)(bb * HEADS + h) * SEQ + s) * HD + d] = f2bf(val);
            }
}

// ===========================================================================
// Kernel 2: flash attention.  One block per (b*h, 64-row q block).
// probs = softmax((QK^T + bias)/sqrt(d) + mask), ctx = probs @ V.
// Online max/sum in LDS; 16 WMMAs per 64-wide k block (32 iterations).
// K tile staged with async global->LDS copies (ASYNCcnt).
// ===========================================================================
__global__ __launch_bounds__(256) void attn_kernel(
    const unsigned short* __restrict__ Qbf, const unsigned short* __restrict__ Kbf,
    const unsigned short* __restrict__ Vbf, const float* __restrict__ bias,
    const float* __restrict__ mask, unsigned short* __restrict__ Ctx)
{
    __shared__ __align__(16) unsigned short ldsK [64  * 136]; // [kRow][d]
    __shared__ __align__(16) unsigned short ldsVt[128 * 72];  // [d][kRow]
    __shared__ __align__(16) unsigned short ldsP [64  * 72];  // [qRow][kRow] bf16
    __shared__ float sM[64], sL[64], sC[64];
    __shared__ float sPMax[2][64], sPSum[2][64];

    const int bh = blockIdx.y;             // 0..31
    const int b  = bh / HEADS;
    const int h  = bh % HEADS;
    const int q0 = blockIdx.x * 64;

    const int tid  = threadIdx.x;
    const int wid  = tid >> 5;
    const int lane = tid & 31;
    const int half = lane >> 4;
    const int l15  = lane & 15;
    const int qS   = wid & 3;              // 16-row q sub-tile
    const int kS   = wid >> 2;             // score phase: 32-col k sub-tile
    const int dS   = wid >> 2;             // PV phase: 64-col d sub-tile

    if (tid < 64) { sM[tid] = -3.0e38f; sL[tid] = 0.0f; }
    __syncthreads();

    // Q fragments (persistent, 4 d-chunks of 32)
    const unsigned short* Qg = Qbf + ((size_t)(b * HEADS + h) * SEQ + q0) * HD;
    FragAB qf[4];
#pragma unroll
    for (int dc = 0; dc < 4; dc++) {
        int mrow = qS * 16 + l15;
#pragma unroll
        for (int i = 0; i < 16; i++) {
            int kk = half * 16 + i;
            qf[dc].u[i] = Qg[(size_t)mrow * HD + dc * 32 + kk];
        }
    }

    FragC o[4];
#pragma unroll
    for (int t = 0; t < 4; t++)
#pragma unroll
        for (int e = 0; e < 8; e++) o[t].f[e] = 0.0f;

    const float rs = 0.08838834764831845f;  // 1/sqrt(128)

    for (int kb = 0; kb < SEQ / 64; kb++) {
        const int kBase = kb * 64;
        const unsigned short* Kg = Kbf + ((size_t)(b * HEADS + h) * SEQ + kBase) * HD;
        const unsigned short* Vg = Vbf + ((size_t)(b * HEADS + h) * SEQ + kBase) * HD;
        if (kb + 1 < SEQ / 64)
            __builtin_prefetch(Kg + (size_t)64 * HD, 0, 1);   // global_prefetch_b8

        // stage K tile 64x128 bf16: async copy, 16B per lane, LDS row pitch
        // 272B (16B aligned)
#pragma unroll
        for (int t = 0; t < 4; t++) {
            int idx = t * 256 + tid;
            int r = idx >> 4, c8 = idx & 15;
            async_copy_b128(Kg + (size_t)r * HD + c8 * 8, &ldsK[r * 136 + c8 * 8]);
        }
        // stage V tile transposed: ldsVt[d][kRow] (needs transpose -> scalar)
#pragma unroll
        for (int t = 0; t < 32; t++) {
            int idx = t * 256 + tid;
            int r = idx >> 7, c = idx & 127;
            ldsVt[c * 72 + r] = Vg[(size_t)r * HD + c];
        }
        async_wait();
        __syncthreads();

        // ---- scores: wave computes 16q x 32k (2 tiles x 4 d-chunks) ----
        FragC sc[2];
#pragma unroll
        for (int tn = 0; tn < 2; tn++)
#pragma unroll
            for (int e = 0; e < 8; e++) sc[tn].f[e] = 0.0f;
#pragma unroll
        for (int tn = 0; tn < 2; tn++) {
            int ncol = kS * 32 + tn * 16 + l15;
#pragma unroll
            for (int dc = 0; dc < 4; dc++) {
                FragAB bk;
#pragma unroll
                for (int i = 0; i < 16; i++)
                    bk.u[i] = ldsK[ncol * 136 + dc * 32 + half * 16 + i];
                sc[tn].v = __builtin_amdgcn_wmma_f32_16x16x32_bf16(
                    false, qf[dc].v, false, bk.v, (short)0, sc[tn].v, false, false);
            }
        }
        // bias before scale, mask after (faithful to reference)
#pragma unroll
        for (int tn = 0; tn < 2; tn++)
#pragma unroll
            for (int e = 0; e < 8; e++) {
                int qr = q0 + qS * 16 + e + 8 * half;
                int kc = kBase + kS * 32 + tn * 16 + l15;
                float bv = bias[((size_t)h * SEQ + qr) * SEQ + kc];
                float mv = mask[((size_t)b * SEQ + qr) * SEQ + kc];
                sc[tn].f[e] = (sc[tn].f[e] + bv) * rs + mv;
            }

        // ---- per-row max within wave (16 lanes hold one row) ----
        float rmax[8];
#pragma unroll
        for (int e = 0; e < 8; e++) rmax[e] = fmaxf(sc[0].f[e], sc[1].f[e]);
#pragma unroll
        for (int off = 1; off < 16; off <<= 1)
#pragma unroll
            for (int e = 0; e < 8; e++)
                rmax[e] = fmaxf(rmax[e], __shfl_xor(rmax[e], off, 32));
        if (l15 == 0)
#pragma unroll
            for (int e = 0; e < 8; e++)
                sPMax[kS][qS * 16 + e + 8 * half] = rmax[e];
        __syncthreads();

        if (tid < 64) {
            float nm = fmaxf(sM[tid], fmaxf(sPMax[0][tid], sPMax[1][tid]));
            sC[tid] = __expf(sM[tid] - nm);
            sM[tid] = nm;
        }
        __syncthreads();

        // ---- exp, P -> LDS (bf16), row partial sums ----
        float mrow[8], rsum[8];
#pragma unroll
        for (int e = 0; e < 8; e++) {
            mrow[e] = sM[qS * 16 + e + 8 * half];
            rsum[e] = 0.0f;
        }
#pragma unroll
        for (int tn = 0; tn < 2; tn++)
#pragma unroll
            for (int e = 0; e < 8; e++) {
                float p = __expf(sc[tn].f[e] - mrow[e]);
                rsum[e] += p;
                ldsP[(qS * 16 + e + 8 * half) * 72 + kS * 32 + tn * 16 + l15] = f2bf(p);
            }
#pragma unroll
        for (int off = 1; off < 16; off <<= 1)
#pragma unroll
            for (int e = 0; e < 8; e++)
                rsum[e] += __shfl_xor(rsum[e], off, 32);
        if (l15 == 0)
#pragma unroll
            for (int e = 0; e < 8; e++)
                sPSum[kS][qS * 16 + e + 8 * half] = rsum[e];
        __syncthreads();

        if (tid < 64)
            sL[tid] = sL[tid] * sC[tid] + sPSum[0][tid] + sPSum[1][tid];

        // ---- rescale O, then O += P @ V (2 k-chunks x 4 d tiles) ----
        float crow[8];
#pragma unroll
        for (int e = 0; e < 8; e++) crow[e] = sC[qS * 16 + e + 8 * half];
#pragma unroll
        for (int t = 0; t < 4; t++)
#pragma unroll
            for (int e = 0; e < 8; e++) o[t].f[e] *= crow[e];

#pragma unroll
        for (int kc2 = 0; kc2 < 2; kc2++) {
            FragAB pa;
            int mrow2 = qS * 16 + l15;
#pragma unroll
            for (int i = 0; i < 16; i++) {
                int kk = half * 16 + i;
                pa.u[i] = ldsP[mrow2 * 72 + kc2 * 32 + kk];
            }
#pragma unroll
            for (int tn = 0; tn < 4; tn++) {
                FragAB vb;
                int ncol = dS * 64 + tn * 16 + l15;
#pragma unroll
                for (int i = 0; i < 16; i++)
                    vb.u[i] = ldsVt[ncol * 72 + kc2 * 32 + half * 16 + i];
                o[tn].v = __builtin_amdgcn_wmma_f32_16x16x32_bf16(
                    false, pa.v, false, vb.v, (short)0, o[tn].v, false, false);
            }
        }
        __syncthreads();
    }

    // ---- finalize: O /= l, store ctx as [b, s, h*128+d] bf16 ----
    float linv[8];
#pragma unroll
    for (int e = 0; e < 8; e++)
        linv[e] = 1.0f / sL[qS * 16 + e + 8 * half];
    unsigned short* Cg = Ctx + ((size_t)b * SEQ + q0) * INNER + h * HD;
#pragma unroll
    for (int tn = 0; tn < 4; tn++)
#pragma unroll
        for (int e = 0; e < 8; e++) {
            int row = qS * 16 + e + 8 * half;
            int d   = dS * 64 + tn * 16 + l15;
            Cg[(size_t)row * INNER + d] = f2bf(o[tn].f[e] * linv[e]);
        }
}

// ===========================================================================
// Kernel 3: output projection.  out = ctx(bf16) @ Wo + bo  (f32 out)
// A tile staged with async global->LDS copies (row pitch 80B, 16B aligned).
// ===========================================================================
__global__ __launch_bounds__(256) void out_proj_kernel(
    const unsigned short* __restrict__ Ctx, const float* __restrict__ Wo,
    const float* __restrict__ bo, float* __restrict__ out)
{
    __shared__ __align__(16) unsigned short ldsA[128 * 40];
    __shared__ __align__(16) unsigned short ldsB[128 * 40];

    const int tid  = threadIdx.x;
    const int wid  = tid >> 5;
    const int lane = tid & 31;
    const int half = lane >> 4;
    const int l15  = lane & 15;

    const int bm = blockIdx.y * 128;
    const int bn = blockIdx.x * 128;
    const int wm = (wid & 3) * 32;
    const int wn = (wid >> 2) * 64;

    FragC acc[2][4];
#pragma unroll
    for (int tm = 0; tm < 2; tm++)
#pragma unroll
        for (int tn = 0; tn < 4; tn++)
#pragma unroll
            for (int e = 0; e < 8; e++) acc[tm][tn].f[e] = 0.0f;

    for (int k0 = 0; k0 < INNER; k0 += 32) {
        // A tile 128x32 bf16 = 512 x 16B chunks, async to LDS
#pragma unroll
        for (int t = 0; t < 2; t++) {
            int idx = t * 256 + tid;
            int r = idx >> 2, c4 = idx & 3;
            async_copy_b128(Ctx + (size_t)(bm + r) * INNER + k0 + c4 * 8,
                            &ldsA[r * 40 + c4 * 8]);
        }
#pragma unroll
        for (int t = 0; t < 16; t++) {
            int idx = t * 256 + tid;
            int r = idx >> 7, c = idx & 127;
            ldsB[c * 40 + r] = f2bf(Wo[(size_t)(k0 + r) * HIDDEN + bn + c]);
        }
        async_wait();
        __syncthreads();

        FragAB a[2], b[4];
#pragma unroll
        for (int tm = 0; tm < 2; tm++) {
            int mrow = wm + tm * 16 + l15;
#pragma unroll
            for (int i = 0; i < 16; i++) {
                int kk = half * 16 + i;
                a[tm].u[i] = ldsA[mrow * 40 + kk];
            }
        }
#pragma unroll
        for (int tn = 0; tn < 4; tn++) {
            int ncol = wn + tn * 16 + l15;
#pragma unroll
            for (int i = 0; i < 16; i++)
                b[tn].u[i] = ldsB[ncol * 40 + half * 16 + i];
        }
#pragma unroll
        for (int tm = 0; tm < 2; tm++)
#pragma unroll
            for (int tn = 0; tn < 4; tn++)
                acc[tm][tn].v = __builtin_amdgcn_wmma_f32_16x16x32_bf16(
                    false, a[tm].v, false, b[tn].v, (short)0, acc[tm][tn].v,
                    false, false);
        __syncthreads();
    }

#pragma unroll
    for (int tm = 0; tm < 2; tm++)
#pragma unroll
        for (int tn = 0; tn < 4; tn++)
#pragma unroll
            for (int e = 0; e < 8; e++) {
                int row = bm + wm + tm * 16 + e + 8 * half;
                int col = bn + wn + tn * 16 + l15;
                out[(size_t)row * HIDDEN + col] = acc[tm][tn].f[e] + bo[col];
            }
}

// ===========================================================================
extern "C" void kernel_launch(void* const* d_in, const int* in_sizes, int n_in,
                              void* d_out, int out_size, void* d_ws, size_t ws_size,
                              hipStream_t stream) {
    const float* q    = (const float*)d_in[0];
    const float* k    = (const float*)d_in[1];
    const float* v    = (const float*)d_in[2];
    const float* mask = (const float*)d_in[3];
    const float* bias = (const float*)d_in[4];
    const float* Wq   = (const float*)d_in[5];
    const float* bq   = (const float*)d_in[6];
    const float* Wk   = (const float*)d_in[7];
    const float* bk   = (const float*)d_in[8];
    const float* Wv   = (const float*)d_in[9];
    const float* bv   = (const float*)d_in[10];
    const float* Wo   = (const float*)d_in[11];
    const float* bo   = (const float*)d_in[12];

    const size_t elems = (size_t)BATCH * SEQ * INNER;   // 8M
    unsigned short* Qbf = (unsigned short*)d_ws;
    unsigned short* Kbf = Qbf + elems;
    unsigned short* Vbf = Kbf + elems;
    unsigned short* Ctx = Vbf + elems;                  // 64 MB total ws

    dim3 gA(INNER / 128, (BATCH * SEQ) / 128, 3);
    qkv_proj_kernel<<<gA, 256, 0, stream>>>(q, k, v, Wq, Wk, Wv, bq, bk, bv,
                                            Qbf, Kbf, Vbf);

    dim3 gB(SEQ / 64, BATCH * HEADS, 1);
    attn_kernel<<<gB, 256, 0, stream>>>(Qbf, Kbf, Vbf, bias, mask, Ctx);

    dim3 gC(HIDDEN / 128, (BATCH * SEQ) / 128, 1);
    out_proj_kernel<<<gC, 256, 0, stream>>>(Ctx, Wo, bo, (float*)d_out);
}
